// Network_7516192768108
// MI455X (gfx1250) — compile-verified, matrix operand order checked by reading, not run on hardware
//
#include <hip/hip_runtime.h>
#include <math.h>

// ---------------- problem constants ----------------
#define BN 4
#define NA 128
#define NPAIR (BN*NA*NA)        // 65536
#define NBASIS 10
#define HRAD 64
#define CHUNK_ATOMS 16
#define CHUNK_PAIRS (CHUNK_ATOMS*NA)   // 2048
#define MAXOUT 2048

typedef __attribute__((ext_vector_type(16))) _Float16 v16h;
typedef __attribute__((ext_vector_type(8)))  float    v8f;

// ---------------- scalar helpers (branch-free fast math) ----------------
// softplus(t) = max(t,0) + log1p(exp(-|t|)); exp arg in [-inf,0] -> no overflow, no branches
__device__ __forceinline__ float softplus5(float x){
  float t = 5.f*x;
  float e = __expf(-fabsf(t));
  return fmaxf(t, 0.f) + __logf(1.f + e);
}
__device__ __forceinline__ float ssp_f(float x){ return (softplus5(x) - 0.69314718056f) * 0.2f; }
__device__ __forceinline__ float sp_f(float x){ return softplus5(x) * 0.2f; }
__device__ __forceinline__ float sigm(float x){ return __builtin_amdgcn_rcpf(1.f + __expf(-x)); }
__device__ __forceinline__ float leaky(float x){ return x > 0.f ? x : 0.2f*x; }

// ---------------- WMMA helpers (CDNA5 16x16x32 f16 -> f32) ----------------
__device__ __forceinline__ v8f wmma16(v16h a, v16h b, v8f c){
  return __builtin_amdgcn_wmma_f32_16x16x32_f16(false, a, false, b, (short)0, c, false, false);
}

// A fragment: lane&15 = row M; hi = lane>>4 selects K halves {0..7,16..23} vs {8..15,24..31}
__device__ __forceinline__ v16h loadA_lds(const _Float16* __restrict__ row, int hi, int kb){
  v16h a; int k0 = kb + hi*8, k1 = kb + 16 + hi*8;
  #pragma unroll
  for (int t = 0; t < 8; ++t){ a[t] = row[k0+t]; a[t+8] = row[k1+t]; }
  return a;
}
// B fragment (mirrored layout): lane&15 = column N; hi selects same K halves
__device__ __forceinline__ v16h loadB_lds(const _Float16* __restrict__ W, int ldw, int col, int hi, int kb){
  v16h b; int k0 = kb + hi*8, k1 = kb + 16 + hi*8;
  #pragma unroll
  for (int t = 0; t < 8; ++t){ b[t] = W[(k0+t)*ldw + col]; b[t+8] = W[(k1+t)*ldw + col]; }
  return b;
}

// ---------------- f32 -> f16 weight pre-conversion ----------------
__global__ void cvt_kernel(const float* __restrict__ src, _Float16* __restrict__ dst, int n){
  int i = blockIdx.x * blockDim.x + threadIdx.x;
  if (i < n) dst[i] = (_Float16)src[i];
}

// ---------------- stage 1: geometry -> Y1, Y2, basis ----------------
__global__ void geom_kernel(const float* __restrict__ geo,
                            float* __restrict__ Y1, float* __restrict__ Y2,
                            float* __restrict__ bas){
  int idx = blockIdx.x * blockDim.x + threadIdx.x;
  if (idx >= NPAIR) return;
  int b = idx / (NA*NA); int rem = idx % (NA*NA); int a = rem / NA; int n = rem % NA;
  float dx = geo[(size_t)(b*NA+n)*3+0] - geo[(size_t)(b*NA+a)*3+0];
  float dy = geo[(size_t)(b*NA+n)*3+1] - geo[(size_t)(b*NA+a)*3+1];
  float dz = geo[(size_t)(b*NA+n)*3+2] - geo[(size_t)(b*NA+a)*3+2];
  float r = sqrtf(dx*dx + dy*dy + dz*dz);
  float inv = __builtin_amdgcn_rcpf(fmaxf(r, 1e-9f));
  float x = dx*inv, y = dy*inv, z = dz*inv;
  const float s3 = 1.7320508075688772f, s15 = 3.872983346207417f, s5 = 2.23606797749979f;
  Y1[(size_t)idx*3+0] = s3*x; Y1[(size_t)idx*3+1] = s3*y; Y1[(size_t)idx*3+2] = s3*z;
  Y2[(size_t)idx*5+0] = s15*x*y;
  Y2[(size_t)idx*5+1] = s15*y*z;
  Y2[(size_t)idx*5+2] = 0.5f*s5*(3.f*z*z - 1.f);
  Y2[(size_t)idx*5+3] = s15*x*z;
  Y2[(size_t)idx*5+4] = 0.5f*s15*(x*x - y*y);
  const float invstep = 0.7142857142857143f; // 1 / (MAX_RAD/NUM_BASIS)
  #pragma unroll
  for (int k = 0; k < NBASIS; ++k){
    float d = r*invstep - (float)k;
    float c = __cosf(1.5707963267948966f * d);
    bas[(size_t)idx*NBASIS + k] = (fabsf(d) < 1.f) ? c*c : 0.f;
  }
}

// ---------------- embedding ----------------
__global__ void embed_kernel(const int* __restrict__ feat, const float* __restrict__ emb,
                             float* __restrict__ f0){
  int idx = blockIdx.x * blockDim.x + threadIdx.x;
  if (idx >= BN*NA) return;
  int e = feat[idx];
  #pragma unroll
  for (int u = 0; u < 16; ++u) f0[(size_t)idx*64 + u] = emb[e*16 + u];
  #pragma unroll
  for (int u = 16; u < 64; ++u) f0[(size_t)idx*64 + u] = 0.f;
}

// ---------------- radial MLP: fused 3-GEMM chain, WMMA ----------------
// block = 128 threads = 4 waves; each wave owns 16 pair-rows; block covers 64 rows.
__global__ __launch_bounds__(128) void radial_kernel(
    const float* __restrict__ basis,       // CHUNK_PAIRS x 10 (chunk-offset)
    const _Float16* __restrict__ w0h,      // 10 x 64 (f16)
    const _Float16* __restrict__ w1h,      // 64 x 64 (f16)
    const _Float16* __restrict__ w2h,      // 64 x out_dim (f16)
    int out_dim, float* __restrict__ rbuf){
  __shared__ _Float16 sW0[32*HRAD];       // K padded 10 -> 32
  __shared__ _Float16 sW1[HRAD*HRAD];
  __shared__ _Float16 sX[4][16*HRAD];     // per-wave activation tile
  __shared__ _Float16 sB[HRAD*16];        // shared w2 column tile
  int tid = threadIdx.x;
  for (int i = tid; i < 32*HRAD; i += 128){
    int k = i >> 6;
    sW0[i] = (k < NBASIS) ? w0h[(k << 6) + (i & 63)] : (_Float16)0.f;
  }
  // vectorized b128 staging of w1 (4096 halves)
  for (int i = tid*8; i < HRAD*HRAD; i += 128*8)
    *reinterpret_cast<int4*>(&sW1[i]) = *reinterpret_cast<const int4*>(&w1h[i]);
  __syncthreads();

  int wave = tid >> 5, lane = tid & 31, ln = lane & 15, hi = lane >> 4;
  int row0 = blockIdx.x*64 + wave*16;
  _Float16* X = sX[wave];

  // GEMM1: X1 = ssp(basis @ w0), K=10 zero-padded to 32
  v16h a0;
  {
    const float* br = basis + (size_t)(row0 + ln) * NBASIS;
    int k0 = hi*8, k1 = 16 + hi*8;
    #pragma unroll
    for (int t = 0; t < 8; ++t){
      a0[t]   = (k0+t < NBASIS) ? (_Float16)br[k0+t] : (_Float16)0.f;
      a0[t+8] = (k1+t < NBASIS) ? (_Float16)br[k1+t] : (_Float16)0.f;
    }
  }
  #pragma unroll
  for (int j = 0; j < 4; ++j){
    v8f c = {};
    v16h b = loadB_lds(sW0, HRAD, j*16 + ln, hi, 0);
    v8f d = wmma16(a0, b, c);
    #pragma unroll
    for (int r = 0; r < 8; ++r) X[(r + 8*hi)*HRAD + j*16 + ln] = (_Float16)ssp_f(d[r]);
  }
  __syncthreads();

  // GEMM2: X2 = ssp(X1 @ w1), K=64 in two WMMA steps
  const _Float16* Xr = X + ln*HRAD;
  v16h a1a = loadA_lds(Xr, hi, 0), a1b = loadA_lds(Xr, hi, 32);
  __syncthreads();
  #pragma unroll
  for (int j = 0; j < 4; ++j){
    v8f c = {};
    v16h bl = loadB_lds(sW1, HRAD, j*16 + ln, hi, 0);
    v16h bh = loadB_lds(sW1, HRAD, j*16 + ln, hi, 32);
    v8f d = wmma16(a1a, bl, c);
    d = wmma16(a1b, bh, d);
    #pragma unroll
    for (int r = 0; r < 8; ++r) X[(r + 8*hi)*HRAD + j*16 + ln] = (_Float16)ssp_f(d[r]);
  }
  __syncthreads();

  // GEMM3: R = X2 @ w2 ; w2 column tile (64x16 f16) staged cooperatively per iteration
  v16h a2a = loadA_lds(Xr, hi, 0), a2b = loadA_lds(Xr, hi, 32);
  int ntile = out_dim >> 4;
  for (int j = 0; j < ntile; ++j){
    __syncthreads();
    {
      int idx = tid*8;                  // 128 threads x 8 halves = 1024 = 64x16 tile
      int k = idx >> 4, c = idx & 15;
      *reinterpret_cast<int4*>(&sB[idx]) =
          *reinterpret_cast<const int4*>(&w2h[(size_t)k*out_dim + j*16 + c]);
    }
    __syncthreads();
    v8f c = {};
    v16h bl = loadB_lds(sB, 16, ln, hi, 0);
    v16h bh = loadB_lds(sB, 16, ln, hi, 32);
    v8f d = wmma16(a2a, bl, c);
    d = wmma16(a2b, bh, d);
    int col = j*16 + ln;
    #pragma unroll
    for (int r = 0; r < 8; ++r)
      rbuf[(size_t)(row0 + r + 8*hi)*out_dim + col] = d[r];
  }
}

// ---------------- equivariant convolution + gating (VALU) ----------------
// one block per output atom; rbuf holds this chunk's per-pair radial weights
__global__ __launch_bounds__(128) void conv_kernel(
    const float* __restrict__ f0_in, const float* __restrict__ f1_in, int has_l1,
    const float* __restrict__ rbuf, int out_dim,
    const float* __restrict__ Y1g, const float* __restrict__ Y2g,
    const float* __restrict__ maskp, int atom0,
    int M0, int M1, int mul0,
    float* __restrict__ f0_out, float* __restrict__ f1_out){
  __shared__ float sf0[NA*16];
  __shared__ float sf1[NA*48];
  __shared__ float sY1[NA*3];
  __shared__ float sY2[NA*5];
  __shared__ float sS[64];
  int tid = threadIdx.x;
  int flat = atom0 + blockIdx.x;
  int b = flat / NA, a = flat % NA;
  const float scale = 0.08838834764831845f;   // 1/sqrt(N)
  size_t pair0 = (size_t)flat * NA;

  for (int i = tid; i < NA; i += 128){
    float mn = maskp[b*NA + i] * scale;
    #pragma unroll
    for (int v = 0; v < 16; ++v) sf0[i*16+v] = f0_in[((size_t)(b*NA+i))*64 + v] * mn;
    if (has_l1){
      #pragma unroll
      for (int v = 0; v < 48; ++v) sf1[i*48+v] = f1_in[((size_t)(b*NA+i))*48 + v] * mn;
    }
    size_t p = pair0 + i;
    #pragma unroll
    for (int m = 0; m < 3; ++m) sY1[i*3+m] = Y1g[p*3+m];
    #pragma unroll
    for (int m = 0; m < 5; ++m) sY2[i*5+m] = Y2g[p*5+m];
  }
  __syncthreads();

  // r_all channel offsets, matching reference path order
  int off = 0;
  int offA = off; off += M0*16;
  int offB = 0; if (has_l1){ offB = off; off += M0*16; }
  int offC = 0, offD = 0, offE = 0, offF = 0;
  if (M1 > 0){
    offC = off; off += M1*16;
    if (has_l1){ offD = off; off += M1*16; offE = off; off += M1*16; offF = off; off += M1*16; }
  }
  const float is3 = 0.5773502691896258f;   // 1/sqrt(3)  (CG 0-1-1, 1-0-1, 1-1-0)
  const float is6 = 0.4082482904638631f;   // 1/sqrt(6)  (CG 1-1-1)
  const float nq  = 6.123724356957945f;    // ||q||_F = sqrt(37.5)
  const float cq  = 1.9364916731037085f / nq;  // (sqrt15/2)/nq
  const float dq  = 1.118033988749895f  / nq;  // (sqrt5/2)/nq

  float acc = 0.f;
  int rloc = blockIdx.x * NA;
  if (tid < M0){
    int u = tid;
    for (int n = 0; n < NA; ++n){
      const float* w = rbuf + (size_t)(rloc + n) * out_dim;
      __builtin_prefetch(rbuf + (size_t)(rloc + n + 1) * out_dim, 0, 0);
      float s1 = 0.f;
      #pragma unroll
      for (int v = 0; v < 16; ++v) s1 += w[offA + u*16 + v] * sf0[n*16+v];
      acc += s1;
      if (has_l1){
        float s2 = 0.f;
        float y0 = sY1[n*3], y1 = sY1[n*3+1], y2 = sY1[n*3+2];
        #pragma unroll
        for (int v = 0; v < 16; ++v){
          const float* f = &sf1[n*48 + v*3];
          float z = (y0*f[0] + y1*f[1] + y2*f[2]) * is3;
          s2 += w[offB + u*16 + v] * z;
        }
        acc += s2;
      }
    }
    sS[u] = acc;
  } else if (M1 > 0 && tid < M0 + 3*M1){
    int q = tid - M0, u = q/3, i = q%3;
    int i1 = (i+1)%3, i2 = (i+2)%3;
    for (int n = 0; n < NA; ++n){
      const float* w = rbuf + (size_t)(rloc + n) * out_dim;
      __builtin_prefetch(rbuf + (size_t)(rloc + n + 1) * out_dim, 0, 0);
      float yi = sY1[n*3+i], yi1 = sY1[n*3+i1], yi2 = sY1[n*3+i2];
      float sC = 0.f;
      #pragma unroll
      for (int v = 0; v < 16; ++v) sC += w[offC + u*16 + v] * sf0[n*16+v];
      acc += is3 * yi * sC;
      if (has_l1){
        float sD = 0.f, sE = 0.f, sF = 0.f;
        float q0 = sY2[n*5], q1 = sY2[n*5+1], q2 = sY2[n*5+2], q3 = sY2[n*5+3], q4 = sY2[n*5+4];
        #pragma unroll
        for (int v = 0; v < 16; ++v){
          const float* f = &sf1[n*48 + v*3];
          sD += w[offD + u*16 + v] * f[i];
          float cr = f[i1]*yi2 - f[i2]*yi1;                // (f x Y1)_i
          sE += w[offE + u*16 + v] * cr;
          float fx = f[0], fy = f[1], fz = f[2];
          float g;
          if (i == 0)      g = q0*cq*fy - q2*dq*fx + q3*cq*fz + q4*cq*fx;
          else if (i == 1) g = q0*cq*fx + q1*cq*fz - q2*dq*fy - q4*cq*fy;
          else             g = q1*cq*fy + q2*dq*(2.f*fz) + q3*cq*fx;
          sF += w[offF + u*16 + v] * g;
        }
        acc += is3*sD + is6*sE + sF;
      }
    }
  }
  __syncthreads();

  float ma = maskp[b*NA + a];
  if (tid < mul0)
    f0_out[((size_t)(b*NA+a))*64 + tid] = sp_f(sS[tid]) * ma;
  if (M1 > 0 && tid >= M0 && tid < M0 + 3*M1){
    int q = tid - M0, u = q/3, i = q%3;
    float gate = sigm(sS[mul0 + u]);
    f1_out[((size_t)(b*NA+a))*48 + u*3 + i] = acc * gate * ma;
  }
}

// ---------------- head: MLP + atom-batchnorm + pooling ----------------
__global__ void head_lin1(const float* __restrict__ x, const float* __restrict__ W1,
                          const float* __restrict__ b1v, float* __restrict__ y){
  int row = blockIdx.x, k = threadIdx.x;
  float s = b1v[k];
  for (int j = 0; j < 64; ++j) s += x[(size_t)row*64 + j] * W1[j*64 + k];
  y[(size_t)row*64 + k] = s;
}
__global__ void head_stats1(const float* __restrict__ y, float* __restrict__ st){
  __shared__ float rs[256], rq[256];
  int n = blockIdx.x, t = threadIdx.x;
  int b = t >> 6, k = t & 63;
  float v = y[((size_t)(b*NA + n))*64 + k];
  rs[t] = v; rq[t] = v*v; __syncthreads();
  for (int s = 128; s > 0; s >>= 1){ if (t < s){ rs[t] += rs[t+s]; rq[t] += rq[t+s]; } __syncthreads(); }
  if (t == 0){ float mu = rs[0]/256.f; st[n] = mu; st[NA+n] = rq[0]/256.f - mu*mu; }
}
__global__ void head_lin2(const float* __restrict__ y, const float* __restrict__ st,
                          const float* __restrict__ g1, const float* __restrict__ be1,
                          const float* __restrict__ W2, const float* __restrict__ b2v,
                          float* __restrict__ y2){
  __shared__ float rs[64];
  int row = blockIdx.x, n = row % NA, k = threadIdx.x;
  float mu = st[n], var = st[NA+n];
  float h = leaky((y[(size_t)row*64 + k] - mu) * rsqrtf(var + 1e-5f) * g1[n] + be1[n]);
  rs[k] = h * W2[k]; __syncthreads();
  for (int s = 32; s > 0; s >>= 1){ if (k < s) rs[k] += rs[k+s]; __syncthreads(); }
  if (k == 0) y2[row] = rs[0] + b2v[0];
}
__global__ void head_stats2(const float* __restrict__ y2, float* __restrict__ st){
  int n = threadIdx.x;
  float s = 0.f, q = 0.f;
  for (int b = 0; b < BN; ++b){ float v = y2[b*NA + n]; s += v; q += v*v; }
  float mu = s/4.f; st[2*NA+n] = mu; st[3*NA+n] = q/4.f - mu*mu;
}
__global__ void head_out(const float* __restrict__ y2, const float* __restrict__ st,
                         const float* __restrict__ g2, const float* __restrict__ be2,
                         const float* __restrict__ maskp, float* __restrict__ out){
  __shared__ float rs[128], rm[128];
  int b = blockIdx.x, n = threadIdx.x;
  float mu = st[2*NA+n], var = st[3*NA+n];
  float h = leaky((y2[b*NA + n] - mu) * rsqrtf(var + 1e-5f) * g2[n] + be2[n]);
  float m = maskp[b*NA + n];
  rs[n] = h*m; rm[n] = m; __syncthreads();
  for (int s = 64; s > 0; s >>= 1){ if (n < s){ rs[n] += rs[n+s]; rm[n] += rm[n+s]; } __syncthreads(); }
  if (n == 0) out[b] = rs[0] / fmaxf(rm[0], 1.f);
}

// ---------------- workspace layout (floats) ----------------
static constexpr size_t OFF_Y1  = 0;                                  // 65536*3
static constexpr size_t OFF_Y2  = OFF_Y1  + (size_t)NPAIR*3;          // 65536*5
static constexpr size_t OFF_BAS = OFF_Y2  + (size_t)NPAIR*5;          // 65536*10
static constexpr size_t OFF_F0A = OFF_BAS + (size_t)NPAIR*NBASIS;
static constexpr size_t OFF_F1A = OFF_F0A + (size_t)BN*NA*64;
static constexpr size_t OFF_F0B = OFF_F1A + (size_t)BN*NA*48;
static constexpr size_t OFF_F1B = OFF_F0B + (size_t)BN*NA*64;
static constexpr size_t OFF_RB  = OFF_F1B + (size_t)BN*NA*48;
static constexpr size_t OFF_YB  = OFF_RB  + (size_t)CHUNK_PAIRS*MAXOUT;
static constexpr size_t OFF_Y2B = OFF_YB  + (size_t)BN*NA*64;
static constexpr size_t OFF_ST  = OFF_Y2B + (size_t)BN*NA;
static constexpr size_t OFF_WH  = OFF_ST  + (size_t)4*NA;             // f16 weight pool (as floats)

extern "C" void kernel_launch(void* const* d_in, const int* in_sizes, int n_in,
                              void* d_out, int out_size, void* d_ws, size_t ws_size,
                              hipStream_t stream) {
  (void)in_sizes; (void)n_in; (void)out_size; (void)ws_size;
  const int*   features = (const int*)  d_in[0];
  const float* geometry = (const float*)d_in[1];
  const float* maskp    = (const float*)d_in[2];
  const float* emb      = (const float*)d_in[3];
  const float* rw0[4] = { (const float*)d_in[4],  (const float*)d_in[7],  (const float*)d_in[10], (const float*)d_in[13] };
  const float* rw1[4] = { (const float*)d_in[5],  (const float*)d_in[8],  (const float*)d_in[11], (const float*)d_in[14] };
  const float* rw2[4] = { (const float*)d_in[6],  (const float*)d_in[9],  (const float*)d_in[12], (const float*)d_in[15] };
  const float* W1h  = (const float*)d_in[16];
  const float* b1h  = (const float*)d_in[17];
  const float* g1h  = (const float*)d_in[18];
  const float* be1h = (const float*)d_in[19];
  const float* W2h  = (const float*)d_in[20];
  const float* b2h  = (const float*)d_in[21];
  const float* g2h  = (const float*)d_in[22];
  const float* be2h = (const float*)d_in[23];

  float* ws  = (float*)d_ws;
  float* Y1  = ws + OFF_Y1;
  float* Y2  = ws + OFF_Y2;
  float* BAS = ws + OFF_BAS;
  float* F0A = ws + OFF_F0A;
  float* F1A = ws + OFF_F1A;
  float* F0B = ws + OFF_F0B;
  float* F1B = ws + OFF_F1B;
  float* RB  = ws + OFF_RB;
  float* YB  = ws + OFF_YB;
  float* Y2B = ws + OFF_Y2B;
  float* ST  = ws + OFF_ST;

  struct LayerCfg { int has_l1, M0, M1, mul0, out_dim; };
  const LayerCfg L[4] = { {0,32,16,16, 768}, {1,32,16,16,2048}, {1,32,16,16,2048}, {1,64, 0,64,2048} };

  // f16 weight pool layout (in halves)
  _Float16* WH = (_Float16*)(ws + OFF_WH);
  _Float16* w0h[4]; _Float16* w1h[4]; _Float16* w2h[4];
  size_t ho = 0;
  for (int l = 0; l < 4; ++l){ w0h[l] = WH + ho; ho += (size_t)NBASIS*HRAD; }      // 640 each
  for (int l = 0; l < 4; ++l){ w1h[l] = WH + ho; ho += (size_t)HRAD*HRAD; }        // 4096 each
  for (int l = 0; l < 4; ++l){ w2h[l] = WH + ho; ho += (size_t)HRAD*L[l].out_dim; }

  // one-time weight conversion (every call; deterministic)
  for (int l = 0; l < 4; ++l){
    int n0 = NBASIS*HRAD, n1 = HRAD*HRAD, n2 = HRAD*L[l].out_dim;
    cvt_kernel<<<(n0+255)/256, 256, 0, stream>>>(rw0[l], w0h[l], n0);
    cvt_kernel<<<(n1+255)/256, 256, 0, stream>>>(rw1[l], w1h[l], n1);
    cvt_kernel<<<(n2+255)/256, 256, 0, stream>>>(rw2[l], w2h[l], n2);
  }

  geom_kernel<<<(NPAIR + 255)/256, 256, 0, stream>>>(geometry, Y1, Y2, BAS);
  embed_kernel<<<(BN*NA + 127)/128, 128, 0, stream>>>(features, emb, F0A);

  float* f0c = F0A; float* f1c = F1A;
  float* f0n = F0B; float* f1n = F1B;
  const int nchunks = (BN*NA) / CHUNK_ATOMS;   // 32
  for (int l = 0; l < 4; ++l){
    for (int c = 0; c < nchunks; ++c){
      radial_kernel<<<CHUNK_PAIRS/64, 128, 0, stream>>>(
          BAS + (size_t)c * CHUNK_PAIRS * NBASIS,
          w0h[l], w1h[l], w2h[l], L[l].out_dim, RB);
      conv_kernel<<<CHUNK_ATOMS, 128, 0, stream>>>(
          f0c, f1c, L[l].has_l1, RB, L[l].out_dim,
          Y1, Y2, maskp, c*CHUNK_ATOMS,
          L[l].M0, L[l].M1, L[l].mul0, f0n, f1n);
    }
    float* t0 = f0c; f0c = f0n; f0n = t0;
    float* t1 = f1c; f1c = f1n; f1n = t1;
  }

  head_lin1  <<<BN*NA, 64,  0, stream>>>(f0c, W1h, b1h, YB);
  head_stats1<<<NA,    256, 0, stream>>>(YB, ST);
  head_lin2  <<<BN*NA, 64,  0, stream>>>(YB, ST, g1h, be1h, W2h, b2h, Y2B);
  head_stats2<<<1,     NA,  0, stream>>>(Y2B, ST);
  head_out   <<<BN,    NA,  0, stream>>>(Y2B, ST, g2h, be2h, maskp, (float*)d_out);
}
